// SelfAttention_79972291051811
// MI455X (gfx1250) — matrix operand order repaired and checked by reading, not compiled, not run on hardware
//
#include <hip/hip_runtime.h>

// ---------------------------------------------------------------------------
// Causal self-attention for MI455X (gfx1250), bf16 WMMA pipeline.
//   Kernel 1: fused QKV projection GEMM (f32 in -> bf16 Q,K + bf16 V^T)
//   Kernel 2: flash-attention (online softmax, WMMA for QK^T and PV)
// ---------------------------------------------------------------------------

typedef __bf16 bf16;
typedef __attribute__((ext_vector_type(16))) bf16  v16bf;
typedef __attribute__((ext_vector_type(8)))  bf16  v8bf;
typedef __attribute__((ext_vector_type(8)))  float v8f;
typedef __attribute__((ext_vector_type(4)))  int   v4i;

// Address-space-qualified int4 pointer types for the async-to-LDS builtin.
typedef __attribute__((address_space(1))) v4i gv4i;   // global
typedef __attribute__((address_space(3))) v4i lv4i;   // LDS

#define B_DIM 4
#define T_DIM 2048
#define C_DIM 1024
#define BT    (B_DIM * T_DIM)   // 8192

#if __has_builtin(__builtin_amdgcn_global_load_async_to_lds_b128)
#define HAVE_ASYNC_LDS 1
#else
#define HAVE_ASYNC_LDS 0
#endif

__device__ __forceinline__ void wait_asynccnt0() {
#if __has_builtin(__builtin_amdgcn_s_wait_asynccnt)
    __builtin_amdgcn_s_wait_asynccnt(0);
#else
    asm volatile("s_wait_asynccnt 0" ::: "memory");
#endif
}

// Async copy of 16 bytes global -> LDS (ASYNCcnt path on gfx1250).
__device__ __forceinline__ void async_copy_b128(const bf16* gsrc, bf16* ldst) {
#if HAVE_ASYNC_LDS
    // AS1 pointer: generic global address == global address on amdgcn.
    gv4i* gp = (gv4i*)(unsigned long long)gsrc;
    // AS3 pointer: low 32 bits of a generic pointer to __shared__ hold the
    // raw LDS offset (aperture bits live in the high half), and AS3 pointers
    // are 32-bit, so truncating is the correct conversion.
    lv4i* lp = (lv4i*)(unsigned int)(unsigned long long)ldst;
    __builtin_amdgcn_global_load_async_to_lds_b128(gp, lp, 0, 0);
#else
    *reinterpret_cast<v8bf*>(ldst) = *reinterpret_cast<const v8bf*>(gsrc);
#endif
}

// Assemble a v16bf A/B fragment from two 16-byte aligned 8-element chunks.
__device__ __forceinline__ v16bf make_frag(const bf16* lo, const bf16* hi) {
    v16bf f;
    *reinterpret_cast<v8bf*>(&f)       = *reinterpret_cast<const v8bf*>(lo);
    *(reinterpret_cast<v8bf*>(&f) + 1) = *reinterpret_cast<const v8bf*>(hi);
    return f;
}

// ---------------------------------------------------------------------------
// Kernel 1: Q = x@Wq+bq, K = x@Wk+bk, V = x@Wv+bv  (all via bf16 WMMA, f32 acc)
// Block: 256 threads (8 waves, 2x4 wave grid). Block tile: 64(M) x 128(N).
// Each wave: 32x32 per weight -> 2x2 WMMA tiles x 3 weights.
// Outputs: Qb[t][c] bf16, Kb[t][c] bf16, Vt[c][t] bf16 (transposed for PV).
// ---------------------------------------------------------------------------
__global__ __launch_bounds__(256)
void qkv_proj_kernel(const float* __restrict__ x,
                     const float* __restrict__ Wq, const float* __restrict__ bq,
                     const float* __restrict__ Wk, const float* __restrict__ bk,
                     const float* __restrict__ Wv, const float* __restrict__ bv,
                     bf16* __restrict__ Qb, bf16* __restrict__ Kb,
                     bf16* __restrict__ Vt)
{
    const int tid   = threadIdx.x;
    const int lane  = tid & 31;
    const int wave  = tid >> 5;
    const int n15   = lane & 15;
    const int hi    = lane >> 4;          // 0: lanes 0-15, 1: lanes 16-31

    const int mb    = blockIdx.x & 127;   // 128 tiles of 64 rows
    const int nb    = blockIdx.x >> 7;    // 8 tiles of 128 cols
    const int mBase = mb * 64;
    const int nBase = nb * 128;
    const int waveM = wave & 1;           // 2 waves along M
    const int waveN = wave >> 1;          // 4 waves along N

    // LDS: bf16 x-tile (row-major, k contiguous) and W-tiles (transposed: [n][k])
    const int XP = 40;                    // 32 + 8 pad -> 80B pitch (16B aligned)
    const int WP = 40;
    __shared__ bf16 Xs [64  * 40];
    __shared__ bf16 Wqs[128 * 40];
    __shared__ bf16 Wks[128 * 40];
    __shared__ bf16 Wvs[128 * 40];

    v8f accQ[2][2], accK[2][2], accV[2][2];
#pragma unroll
    for (int mi = 0; mi < 2; ++mi)
#pragma unroll
        for (int ni = 0; ni < 2; ++ni) {
            accQ[mi][ni] = v8f{};
            accK[mi][ni] = v8f{};
            accV[mi][ni] = v8f{};
        }

    for (int kb = 0; kb < C_DIM / 32; ++kb) {
        // ---- stage x tile: 64 x 32 f32 -> bf16 ----
#pragma unroll
        for (int i = 0; i < 8; ++i) {
            int e = tid + 256 * i;        // 2048 elements
            int r = e >> 5, k = e & 31;
            Xs[r * XP + k] = (bf16)x[(size_t)(mBase + r) * C_DIM + kb * 32 + k];
        }
        // ---- stage W tiles: 32 x 128 f32 -> bf16, transposed to [n][k] ----
#pragma unroll
        for (int i = 0; i < 16; ++i) {
            int e = tid + 256 * i;        // 4096 elements per weight
            int k = e >> 7, n = e & 127;
            size_t gi = (size_t)(kb * 32 + k) * C_DIM + nBase + n;
            Wqs[n * WP + k] = (bf16)Wq[gi];
            Wks[n * WP + k] = (bf16)Wk[gi];
            Wvs[n * WP + k] = (bf16)Wv[gi];
        }
        __syncthreads();

        // ---- fragments ----
        v16bf aF[2], bQ[2], bK[2], bV[2];
#pragma unroll
        for (int mi = 0; mi < 2; ++mi) {
            const bf16* rowp = &Xs[(waveM * 32 + mi * 16 + n15) * XP];
            // 16-bit A layout: lanes<16 -> K 0-7 & 16-23; lanes>=16 -> 8-15 & 24-31
            aF[mi] = make_frag(rowp + (hi ? 8 : 0), rowp + (hi ? 24 : 16));
        }
#pragma unroll
        for (int ni = 0; ni < 2; ++ni) {
            const int nrow = (waveN * 32 + ni * 16 + n15) * WP + (hi ? 8 : 0);
            bQ[ni] = make_frag(&Wqs[nrow], &Wqs[nrow] + 16);
            bK[ni] = make_frag(&Wks[nrow], &Wks[nrow] + 16);
            bV[ni] = make_frag(&Wvs[nrow], &Wvs[nrow] + 16);
        }
#pragma unroll
        for (int mi = 0; mi < 2; ++mi)
#pragma unroll
            for (int ni = 0; ni < 2; ++ni) {
                accQ[mi][ni] = __builtin_amdgcn_wmma_f32_16x16x32_bf16(
                    false, aF[mi], false, bQ[ni], (short)0, accQ[mi][ni], false, false);
                accK[mi][ni] = __builtin_amdgcn_wmma_f32_16x16x32_bf16(
                    false, aF[mi], false, bK[ni], (short)0, accK[mi][ni], false, false);
                accV[mi][ni] = __builtin_amdgcn_wmma_f32_16x16x32_bf16(
                    false, aF[mi], false, bV[ni], (short)0, accV[mi][ni], false, false);
            }
        __syncthreads();
    }

    // ---- epilogue: bias add + store (Q,K row-major bf16; V transposed) ----
#pragma unroll
    for (int ni = 0; ni < 2; ++ni) {
        const int c = nBase + waveN * 32 + ni * 16 + n15;
        const float vbq = bq[c], vbk = bk[c], vbv = bv[c];
#pragma unroll
        for (int mi = 0; mi < 2; ++mi) {
            const int tBase = mBase + waveM * 32 + mi * 16 + 8 * hi; // M = r + 8*hi
            v8bf vpack;
#pragma unroll
            for (int r = 0; r < 8; ++r) {
                const size_t t = (size_t)(tBase + r);
                Qb[t * C_DIM + c] = (bf16)(accQ[mi][ni][r] + vbq);
                Kb[t * C_DIM + c] = (bf16)(accK[mi][ni][r] + vbk);
                vpack[r] = (bf16)(accV[mi][ni][r] + vbv);
            }
            *reinterpret_cast<v8bf*>(Vt + (size_t)c * BT + tBase) = vpack;
        }
    }
}

// ---------------------------------------------------------------------------
// Kernel 2: flash attention. One block = 16 query rows of one batch.
// 8 waves: in the S stage each wave owns 16 of the 128 keys in the block;
// in the PV stage each wave owns a 128-wide slice of the C=1024 output dim.
// ---------------------------------------------------------------------------
__global__ __launch_bounds__(256)
void flash_attn_kernel(const bf16* __restrict__ Qb, const bf16* __restrict__ Kb,
                       const bf16* __restrict__ Vt, float* __restrict__ out)
{
    const int tid  = threadIdx.x;
    const int lane = tid & 31;
    const int w    = tid >> 5;
    const int n15  = lane & 15;
    const int hi   = lane >> 4;

    const int b      = blockIdx.x >> 7;      // 128 q-tiles per batch
    const int qt     = blockIdx.x & 127;
    const int qBase  = qt * 16;
    const size_t rowOff = (size_t)b * T_DIM; // batch row offset into Q/K/Vt columns
    const float scale = 0.03125f;            // 1/sqrt(1024)

    const int QP = 1032;                     // 1024 + 8 pad
    const int PP = 136;                      // 128 + 8 pad
    __shared__ bf16  Qs[16 * 1032];          // 33 KB
    __shared__ bf16  Pb[16 * 136];           // 4.25 KB
    __shared__ float s_m[16], s_l[16], s_mnew[16], s_corr[16];
    __shared__ float s_tmax[8][16], s_tsum[8][16];

    // ---- stage Q strip (16 x 1024 bf16): pure byte copy -> async-to-LDS ----
#pragma unroll
    for (int i = 0; i < 8; ++i) {
        int chunk = tid + 256 * i;           // 2048 chunks of 8 bf16
        int r  = chunk >> 7;
        int cc = (chunk & 127) * 8;
        async_copy_b128(&Qb[(rowOff + qBase + r) * C_DIM + cc], &Qs[r * QP + cc]);
    }
    if (tid < 16) { s_m[tid] = -1e30f; s_l[tid] = 0.0f; }
#if HAVE_ASYNC_LDS
    wait_asynccnt0();
#endif
    __syncthreads();

    v8f O[8];
#pragma unroll
    for (int ct = 0; ct < 8; ++ct) O[ct] = v8f{};

    const int sbMax = (qBase + 15) >> 7;     // causal clip: s <= t only
    for (int sb = 0; sb <= sbMax; ++sb) {
        const int sBase = sb * 128;

        // ======== S = (Q K^T) tile: 16 x 16, contraction over C=1024 ========
        v8f S = v8f{};
        const int  sCol  = sBase + w * 16 + n15;     // this lane's key row (B N index)
        const bf16* krow = &Kb[(rowOff + sCol) * C_DIM + (hi ? 8 : 0)];
        const bf16* qrow = &Qs[n15 * QP];
#pragma unroll 8
        for (int kc = 0; kc < 32; ++kc) {
            v16bf aF = make_frag(qrow + kc * 32 + (hi ? 8 : 0),
                                 qrow + kc * 32 + (hi ? 24 : 16));
            v16bf bF = make_frag(krow + kc * 32, krow + kc * 32 + 16);
            S = __builtin_amdgcn_wmma_f32_16x16x32_bf16(
                    false, aF, false, bF, (short)0, S, false, false);
        }

        // ---- scale + causal mask + per-row max (reduce over 16-lane half) ----
        float mx[8];
#pragma unroll
        for (int r = 0; r < 8; ++r) {
            const int t = qBase + r + 8 * hi;        // global query row
            float v = (sCol <= t) ? S[r] * scale : -1e30f;
            S[r] = v;
            v = fmaxf(v, __shfl_xor(v, 1, 32));
            v = fmaxf(v, __shfl_xor(v, 2, 32));
            v = fmaxf(v, __shfl_xor(v, 4, 32));
            v = fmaxf(v, __shfl_xor(v, 8, 32));
            mx[r] = v;
        }
        if (n15 == 0) {
#pragma unroll
            for (int r = 0; r < 8; ++r) s_tmax[w][r + 8 * hi] = mx[r];
        }
        __syncthreads();

        // ---- block-wide running max / correction (one thread per row) ----
        if (tid < 16) {
            float mOld = s_m[tid];
            float mT = s_tmax[0][tid];
#pragma unroll
            for (int ww = 1; ww < 8; ++ww) mT = fmaxf(mT, s_tmax[ww][tid]);
            float mNew = fmaxf(mOld, mT);
            s_mnew[tid] = mNew;
            s_corr[tid] = __expf(mOld - mNew);
            s_m[tid]    = mNew;
        }
        __syncthreads();

        // ---- P = exp(S - m), row sums, stage P to LDS, rescale O ----
#pragma unroll
        for (int r = 0; r < 8; ++r) {
            const int row = r + 8 * hi;
            float p = __expf(S[r] - s_mnew[row]);
            Pb[row * PP + w * 16 + n15] = (bf16)p;
            float sv = p;
            sv += __shfl_xor(sv, 1, 32);
            sv += __shfl_xor(sv, 2, 32);
            sv += __shfl_xor(sv, 4, 32);
            sv += __shfl_xor(sv, 8, 32);
            if (n15 == 0) s_tsum[w][row] = sv;
            const float corr = s_corr[row];
#pragma unroll
            for (int ct = 0; ct < 8; ++ct) O[ct][r] *= corr;
        }
        __syncthreads();

        if (tid < 16) {
            float acc = s_l[tid] * s_corr[tid];
#pragma unroll
            for (int ww = 0; ww < 8; ++ww) acc += s_tsum[ww][tid];
            s_l[tid] = acc;
        }

        // ======== O += P V : contraction over the 128 keys of this block ====
        v16bf pF[4];
        const bf16* prow = &Pb[n15 * PP];
#pragma unroll
        for (int ks = 0; ks < 4; ++ks)
            pF[ks] = make_frag(prow + ks * 32 + (hi ? 8 : 0),
                               prow + ks * 32 + (hi ? 24 : 16));
#pragma unroll
        for (int ct = 0; ct < 8; ++ct) {
            const int c = w * 128 + ct * 16 + n15;   // lane's output column (B N index)
            const bf16* vcol = &Vt[(size_t)c * BT + rowOff + sBase + (hi ? 8 : 0)];
#pragma unroll
            for (int ks = 0; ks < 4; ++ks) {
                v16bf bF = make_frag(vcol + ks * 32, vcol + ks * 32 + 16);
                O[ct] = __builtin_amdgcn_wmma_f32_16x16x32_bf16(
                            false, pF[ks], false, bF, (short)0, O[ct], false, false);
            }
        }

        // prefetch next K block into cache (global_prefetch_b8 on gfx1250)
        if (sb < sbMax) {
            const bf16* nk = &Kb[(rowOff + sBase + 128 + (tid >> 1)) * C_DIM
                                 + (size_t)(tid & 1) * 512];
            __builtin_prefetch(nk, 0, 1);
        }
        __syncthreads();
    }

    // ---- normalize and store fp32 output ----
#pragma unroll
    for (int r = 0; r < 8; ++r) {
        const int row = r + 8 * hi;
        const float inv = 1.0f / s_l[row];
        float* orow = &out[(rowOff + qBase + row) * C_DIM];
#pragma unroll
        for (int ct = 0; ct < 8; ++ct)
            orow[w * 128 + ct * 16 + n15] = O[ct][r] * inv;
    }
}

// ---------------------------------------------------------------------------
// Launcher
// ---------------------------------------------------------------------------
extern "C" void kernel_launch(void* const* d_in, const int* in_sizes, int n_in,
                              void* d_out, int out_size, void* d_ws, size_t ws_size,
                              hipStream_t stream) {
    const float* x  = (const float*)d_in[0];
    const float* Wq = (const float*)d_in[1];
    const float* bq = (const float*)d_in[2];
    const float* bk_unused = nullptr; (void)bk_unused;
    const float* Wk = (const float*)d_in[3];
    const float* bk = (const float*)d_in[4];
    const float* Wv = (const float*)d_in[5];
    const float* bv = (const float*)d_in[6];
    float* out = (float*)d_out;

    // Workspace layout: Qb | Kb | Vt (each BT*C bf16 = 16 MB; total 48 MB)
    bf16* Qb = (bf16*)d_ws;
    bf16* Kb = Qb + (size_t)BT * C_DIM;
    bf16* Vt = Kb + (size_t)BT * C_DIM;

    // Kernel 1: 128 (M/64) x 8 (N/128) = 1024 blocks, 256 threads.
    qkv_proj_kernel<<<1024, 256, 0, stream>>>(x, Wq, bq, Wk, bk, Wv, bv,
                                              Qb, Kb, Vt);
    // Kernel 2: B * T/16 = 512 blocks, 256 threads.
    flash_attn_kernel<<<512, 256, 0, stream>>>(Qb, Kb, Vt, out);
}
